// MultiHeadAttention_40269613367608
// MI455X (gfx1250) — matrix operand order repaired and checked by reading, not compiled, not run on hardware
//
#include <hip/hip_runtime.h>

typedef __attribute__((ext_vector_type(16))) _Float16 v16h;
typedef __attribute__((ext_vector_type(8)))  float    v8f;
typedef unsigned int u32x4 __attribute__((ext_vector_type(4)));
typedef int          i32x4 __attribute__((ext_vector_type(4)));
typedef int          i32x8 __attribute__((ext_vector_type(8)));

#define WMMA_F16(a,b,c) __builtin_amdgcn_wmma_f32_16x16x32_f16(false,(a),false,(b),(short)0,(c),false,false)
#define WAIT_ASYNCCNT(imm) asm volatile("s_wait_asynccnt " #imm ::: "memory")

constexpr int Bb = 4, Tt = 2048, Cc = 1024, Hh = 16, Dd = 64;
constexpr int Mrows = Bb * Tt; // 8192

// Low 32 bits of a generic pointer into LDS == byte offset from the wave's
// LDS base (AMDGPU local->flat addrspacecast keeps the offset in the low word).
__device__ inline uint32_t lds_off32(const void* p) {
  return (uint32_t)(uintptr_t)p;
}

// GLOBAL_LOAD_ASYNC_TO_LDS_B128: vdst = LDS byte offset, vaddr = 64-bit global.
__device__ inline void async_b128(uint32_t ldsoff, const void* g) {
  asm volatile("global_load_async_to_lds_b128 %0, %1, off"
               :: "v"(ldsoff), "v"((unsigned long long)(uintptr_t)g)
               : "memory");
}

// TDM 2D tile load (ISA 8.3/8.4): tile tw x th elements (2B), row stride in
// elements. LDS pad codes: padint (0=2DW..4=32DW interval), padamt (3 = 4 DWORDs).
__device__ inline void tdm_load_2d(uint32_t lds_addr, const void* g,
                                   uint32_t tw, uint32_t th,
                                   uint32_t stride_elems,
                                   uint32_t padint, uint32_t padamt) {
  const uint64_t ga = (uint64_t)(uintptr_t)g;
  u32x4 g0;
  g0[0] = 1u;                                   // count=1, user descriptor
  g0[1] = lds_addr;                             // lds_addr
  g0[2] = (uint32_t)ga;                         // global_addr[31:0]
  g0[3] = (uint32_t)(ga >> 32) | (2u << 30);    // global_addr[56:32] | type=2
  i32x8 g1;
  g1[0] = (int)((1u << 16) | (1u << 20) | (padint << 22) | (padamt << 25));
  g1[1] = (int)(tw << 16);                      // tensor_dim0[15:0] @ bits 63:48
  g1[2] = (int)((tw >> 16) | (th << 16));       // tensor_dim0[31:16] | tensor_dim1[15:0]
  g1[3] = (int)((th >> 16) | (tw << 16));       // tensor_dim1[31:16] | tile_dim0
  g1[4] = (int)(th & 0xFFFF);                   // tile_dim1 (tile_dim2 = 0)
  g1[5] = (int)stride_elems;                    // tensor_dim0_stride[31:0]
  g1[6] = 0;
  g1[7] = 0;
  const i32x4 z = {0, 0, 0, 0};
  asm volatile("tensor_load_to_lds %0, %1, %2, %3"
               :: "s"(g0), "s"(g1), "s"(z), "s"(z) : "memory");
}

// ---------------------------------------------------------------------------
__global__ void cvt_f32_f16(const float* __restrict__ s, _Float16* __restrict__ d, int n) {
  int i = blockIdx.x * blockDim.x + threadIdx.x;
  if (i < n) d[i] = (_Float16)s[i];
}

// ---------------------------------------------------------------------------
// out[M,N] = A[M,K] * W[N,K]^T ; M=8192, N=K=1024. BK=64 tiles staged via TDM.
// MODE 0: f16 row-major; MODE 1: f16 transposed (B,H,D,T) (V); MODE 2: f32 -> d_out.
template <int MODE>
__global__ __launch_bounds__(256) void gemm_qkvo(const _Float16* __restrict__ A,
                                                 const _Float16* __restrict__ W,
                                                 void* __restrict__ outv) {
  __shared__ _Float16 As[2][128][72];   // 64 used + 8 pad (TDM pads rows)
  __shared__ _Float16 Ws[2][64][72];
  const int nb   = Cc / 64;                    // 16 col-blocks
  const int bm   = (blockIdx.x / nb) * 128;
  const int bn   = (blockIdx.x % nb) * 64;
  const int wave = threadIdx.x >> 5;
  const int lane = threadIdx.x & 31;
  const int rl   = (wave >> 1) * 32;           // local row base in As
  const int cl   = (wave & 1) * 32;            // local col base in Ws

  auto stage = [&](int kt, int bufi) {
    const int k0 = kt * 64;
    tdm_load_2d(lds_off32(&As[bufi][0][0]), A + (size_t)bm * Cc + k0, 64, 128, Cc, 4, 3);
    tdm_load_2d(lds_off32(&Ws[bufi][0][0]), W + (size_t)bn * Cc + k0, 64, 64,  Cc, 4, 3);
  };

  if (wave == 0) stage(0, 0);

  v8f acc00 = {}, acc01 = {}, acc10 = {}, acc11 = {};
  const int m   = lane & 15;
  const int kbA = (lane & 16) ? 8 : 0;   // A-layout K base
  const int kbB = (lane & 16) ? 16 : 0;  // contiguous k=16h+s base (unused)

  for (int kt = 0; kt < Cc / 64; ++kt) {
    const int bufi = kt & 1;
    if (wave == 0) {
      if (kt + 1 < Cc / 64) {
        stage(kt + 1, bufi ^ 1);
        __builtin_amdgcn_s_wait_tensorcnt((short)2);
      } else {
        __builtin_amdgcn_s_wait_tensorcnt((short)0);
      }
    }
    __syncthreads();

#pragma unroll
    for (int c = 0; c < 2; ++c) {        // two K=32 chunks within the BK=64 tile
      const int kc = c * 32;
      v16h a0, a1, b0, b1;
#pragma unroll
      for (int i = 0; i < 8; ++i) {
        a0[i]     = As[bufi][rl + m][kc + kbA + i];
        a0[i + 8] = As[bufi][rl + m][kc + kbA + 16 + i];
        a1[i]     = As[bufi][rl + 16 + m][kc + kbA + i];
        a1[i + 8] = As[bufi][rl + 16 + m][kc + kbA + 16 + i];
      }
#pragma unroll
      for (int i = 0; i < 8; ++i) {
        b0[i]     = Ws[bufi][cl + m][kc + kbA + i];
        b0[i + 8] = Ws[bufi][cl + m][kc + kbA + 16 + i];
        b1[i]     = Ws[bufi][cl + 16 + m][kc + kbA + i];
        b1[i + 8] = Ws[bufi][cl + 16 + m][kc + kbA + 16 + i];
      }
      acc00 = WMMA_F16(a0, b0, acc00);
      acc01 = WMMA_F16(a0, b1, acc01);
      acc10 = WMMA_F16(a1, b0, acc10);
      acc11 = WMMA_F16(a1, b1, acc11);
    }
    __syncthreads();
  }

  const int cn = lane & 15;
  const int rb = (lane & 16) ? 8 : 0;
#pragma unroll
  for (int mi = 0; mi < 2; ++mi) {
#pragma unroll
    for (int ni = 0; ni < 2; ++ni) {
      const v8f& a = (mi == 0) ? ((ni == 0) ? acc00 : acc01)
                               : ((ni == 0) ? acc10 : acc11);
#pragma unroll
      for (int r = 0; r < 8; ++r) {
        const int mg = bm + rl + mi * 16 + rb + r;
        const int ng = bn + cl + ni * 16 + cn;
        const float val = a[r];
        if (MODE == 0) {
          ((_Float16*)outv)[(size_t)mg * Cc + ng] = (_Float16)val;
        } else if (MODE == 1) {
          const int b = mg >> 11, t = mg & (Tt - 1);
          const int h = ng >> 6,  d = ng & (Dd - 1);
          ((_Float16*)outv)[(((size_t)b * Hh + h) * Dd + d) * Tt + t] = (_Float16)val;
        } else {
          ((float*)outv)[(size_t)mg * Cc + ng] = val;
        }
      }
    }
  }
}

// ---------------------------------------------------------------------------
// RoPE + RMSNorm over D=64, one wave per (b,t,h); 1/sqrt(D) folded into q.
__global__ __launch_bounds__(256) void rope_rms(_Float16* __restrict__ qk,
                                                const float* __restrict__ cosb,
                                                const float* __restrict__ sinb,
                                                float outscale) {
  const int wave = threadIdx.x >> 5;
  const int lane = threadIdx.x & 31;
  const long row = (long)blockIdx.x * 8 + wave;     // 0 .. B*T*H-1
  const int t = (int)((row >> 4) & (Tt - 1));
  _Float16* p = qk + row * Dd;
  const float x1 = (float)p[lane];
  const float x2 = (float)p[lane + 32];
  const float c  = cosb[t * 32 + lane];
  const float s  = sinb[t * 32 + lane];
  const float r1 = x1 * c + x2 * s;
  const float r2 = -x1 * s + x2 * c;
  float ss = r1 * r1 + r2 * r2;
#pragma unroll
  for (int mm = 1; mm < 32; mm <<= 1) ss += __shfl_xor(ss, mm, 32);
  const float inv = rsqrtf(ss * (1.0f / 64.0f) + 1.1920929e-07f) * outscale;
  p[lane]      = (_Float16)(r1 * inv);
  p[lane + 32] = (_Float16)(r2 * inv);
}

// ---------------------------------------------------------------------------
// Flash attention without max-tracking: q,k are RMS-normalized so
// |score| = |q.k|/sqrt(D) <= 8  ->  exp(score) in [3.4e-4, 2981]: f32-safe.
// Per-lane partial row sums; single cross-lane reduction at the end.
// grid = B*H*(T/128), 8 waves/block, 16 q-rows per wave.
__global__ __launch_bounds__(256) void flash_attn(const _Float16* __restrict__ q,
                                                  const _Float16* __restrict__ k,
                                                  const _Float16* __restrict__ vT,
                                                  _Float16* __restrict__ y) {
  __shared__ _Float16 kbuf[2][32][72];  // 64 used + 8 pad
  __shared__ _Float16 vbuf[2][64][40];  // 32 used + 8 pad
  __shared__ _Float16 plds[8][16 * 32]; // per-wave P staging
  const int tid  = threadIdx.x;
  const int wave = tid >> 5;
  const int lane = tid & 31;
  const int qb = blockIdx.x & 15;
  const int h  = (blockIdx.x >> 4) & 15;
  const int b  = blockIdx.x >> 8;
  const int t0 = qb * 128 + wave * 16;

  const _Float16* qp  = q  + (size_t)b * Tt * Cc + h * Dd;   // row stride Cc
  const _Float16* kp  = k  + (size_t)b * Tt * Cc + h * Dd;
  const _Float16* vtp = vT + ((size_t)b * Hh + h) * Dd * Tt; // row stride Tt

  auto stage_kv = [&](int s0, int bufi) {
    { const int r = tid >> 3, ch = (tid & 7) * 8;   // 32 rows x 128B
      async_b128(lds_off32(&kbuf[bufi][r][ch]), kp + (size_t)(s0 + r) * Cc + ch); }
    { const int d = tid >> 2, ch = (tid & 3) * 8;   // 64 rows x 64B
      async_b128(lds_off32(&vbuf[bufi][d][ch]), vtp + (size_t)d * Tt + s0 + ch); }
  };

  // q fragments direct from global (one-time)
  v16h qa0, qa1;
  {
    const int m  = lane & 15;
    const int kb = (lane & 16) ? 8 : 0;
    const _Float16* r0 = qp + (size_t)(t0 + m) * Cc;
#pragma unroll
    for (int i = 0; i < 8; ++i) {
      qa0[i] = r0[kb + i];      qa0[i + 8] = r0[kb + 16 + i];
      qa1[i] = r0[32 + kb + i]; qa1[i + 8] = r0[32 + kb + 16 + i];
    }
  }

  stage_kv(0, 0);

  v8f o0 = {}, o1 = {}, o2 = {}, o3 = {};
  float lsum[8];
#pragma unroll
  for (int r = 0; r < 8; ++r) lsum[r] = 0.0f;

  const int prow = (lane & 16) ? 8 : 0;
  const int nn   = lane & 15;
  const int kbbA = (lane & 16) ? 8 : 0;
  const int kbbB = (lane & 16) ? 16 : 0;

  for (int it = 0; it < Tt / 32; ++it) {
    const int bufi = it & 1;
    if (it + 1 < Tt / 32) {
      stage_kv((it + 1) * 32, bufi ^ 1);
      WAIT_ASYNCCNT(0x2);
    } else {
      WAIT_ASYNCCNT(0x0);
    }
    __syncthreads();

    // K fragments from LDS (B-layout: col = local s row, contiguous d)
    v16h kb0, kb1, kb2, kb3;
#pragma unroll
    for (int i = 0; i < 8; ++i) {
      kb0[i] = kbuf[bufi][nn][kbbA + i];           kb0[i + 8] = kbuf[bufi][nn][kbbA + 16 + i];
      kb1[i] = kbuf[bufi][nn][32 + kbbA + i];      kb1[i + 8] = kbuf[bufi][nn][32 + kbbA + 16 + i];
      kb2[i] = kbuf[bufi][16 + nn][kbbA + i];      kb2[i + 8] = kbuf[bufi][16 + nn][kbbA + 16 + i];
      kb3[i] = kbuf[bufi][16 + nn][32 + kbbA + i]; kb3[i + 8] = kbuf[bufi][16 + nn][32 + kbbA + 16 + i];
    }
    v8f sa = {}, sb = {};
    sa = WMMA_F16(qa0, kb0, sa);
    sa = WMMA_F16(qa1, kb1, sa);
    sb = WMMA_F16(qa0, kb2, sb);
    sb = WMMA_F16(qa1, kb3, sb);

    // exp without shift (bounded scores); accumulate per-lane partial row sums
#pragma unroll
    for (int r = 0; r < 8; ++r) {
      const float p0 = __expf(sa[r]);
      const float p1 = __expf(sb[r]);
      lsum[r] += p0 + p1;
      plds[wave][(prow + r) * 32 + nn]      = (_Float16)p0;
      plds[wave][(prow + r) * 32 + 16 + nn] = (_Float16)p1;
    }

    // reload P as an A-fragment (16x32 over s)
    v16h pa;
#pragma unroll
    for (int i = 0; i < 8; ++i) {
      pa[i]     = plds[wave][nn * 32 + kbbA + i];
      pa[i + 8] = plds[wave][nn * 32 + kbbA + 16 + i];
    }

    // V fragments from LDS (B-layout: col = d, contiguous local s)
    v16h vb0, vb1, vb2, vb3;
#pragma unroll
    for (int i = 0; i < 8; ++i) {
      vb0[i] = vbuf[bufi][nn][kbbA + i];      vb0[i + 8] = vbuf[bufi][nn][kbbA + 16 + i];
      vb1[i] = vbuf[bufi][16 + nn][kbbA + i]; vb1[i + 8] = vbuf[bufi][16 + nn][kbbA + 16 + i];
      vb2[i] = vbuf[bufi][32 + nn][kbbA + i]; vb2[i + 8] = vbuf[bufi][32 + nn][kbbA + 16 + i];
      vb3[i] = vbuf[bufi][48 + nn][kbbA + i]; vb3[i + 8] = vbuf[bufi][48 + nn][kbbA + 16 + i];
    }
    o0 = WMMA_F16(pa, vb0, o0);
    o1 = WMMA_F16(pa, vb1, o1);
    o2 = WMMA_F16(pa, vb2, o2);
    o3 = WMMA_F16(pa, vb3, o3);
    __syncthreads();
  }

  // single end-of-kernel row-sum reduction + normalize + store
#pragma unroll
  for (int r = 0; r < 8; ++r) {
    float rs = lsum[r];
#pragma unroll
    for (int mm = 1; mm < 16; mm <<= 1) rs += __shfl_xor(rs, mm, 32);
    const float inv = 1.0f / rs;
    const int t = t0 + prow + r;
    _Float16* yp = y + (size_t)b * Tt * Cc + (size_t)t * Cc + h * Dd;
    yp[nn]      = (_Float16)(o0[r] * inv);
    yp[16 + nn] = (_Float16)(o1[r] * inv);
    yp[32 + nn] = (_Float16)(o2[r] * inv);
    yp[48 + nn] = (_Float16)(o3[r] * inv);
  }
}

// ---------------------------------------------------------------------------
extern "C" void kernel_launch(void* const* d_in, const int* in_sizes, int n_in,
                              void* d_out, int out_size, void* d_ws, size_t ws_size,
                              hipStream_t stream) {
  (void)in_sizes; (void)n_in; (void)out_size; (void)ws_size;
  const float* x    = (const float*)d_in[0];
  const float* cosb = (const float*)d_in[1];
  const float* sinb = (const float*)d_in[2];
  const float* Wq   = (const float*)d_in[3];
  const float* Wk   = (const float*)d_in[4];
  const float* Wv   = (const float*)d_in[5];
  const float* Wo   = (const float*)d_in[6];

  const size_t NX = (size_t)Mrows * Cc;  // 8,388,608
  const size_t NW = (size_t)Cc * Cc;     // 1,048,576

  _Float16* ws  = (_Float16*)d_ws;
  _Float16* xh  = ws;
  _Float16* wqh = xh  + NX;
  _Float16* wkh = wqh + NW;
  _Float16* wvh = wkh + NW;
  _Float16* woh = wvh + NW;
  _Float16* qh  = woh + NW;
  _Float16* kh  = qh  + NX;
  _Float16* vth = kh  + NX;   // (B,H,D,T)
  _Float16* yh  = vth + NX;

  cvt_f32_f16<<<(int)((NX + 255) / 256), 256, 0, stream>>>(x,  xh,  (int)NX);
  cvt_f32_f16<<<(int)((NW + 255) / 256), 256, 0, stream>>>(Wq, wqh, (int)NW);
  cvt_f32_f16<<<(int)((NW + 255) / 256), 256, 0, stream>>>(Wk, wkh, (int)NW);
  cvt_f32_f16<<<(int)((NW + 255) / 256), 256, 0, stream>>>(Wv, wvh, (int)NW);
  cvt_f32_f16<<<(int)((NW + 255) / 256), 256, 0, stream>>>(Wo, woh, (int)NW);

  dim3 g(1024), blk(256);
  gemm_qkvo<0><<<g, blk, 0, stream>>>(xh, wqh, (void*)qh);
  gemm_qkvo<0><<<g, blk, 0, stream>>>(xh, wkh, (void*)kh);
  gemm_qkvo<1><<<g, blk, 0, stream>>>(xh, wvh, (void*)vth);

  rope_rms<<<16384, 256, 0, stream>>>(qh, cosb, sinb, 0.125f); // fold 1/sqrt(D)
  rope_rms<<<16384, 256, 0, stream>>>(kh, cosb, sinb, 1.0f);

  flash_attn<<<1024, 256, 0, stream>>>(qh, kh, vth, yh);

  gemm_qkvo<2><<<g, blk, 0, stream>>>(yh, woh, d_out);
}
